// CaptionLoss_24962349924617
// MI455X (gfx1250) — matrix operand order repaired and checked by reading, not compiled, not run on hardware
//
#include <hip/hip_runtime.h>
#include <hip/hip_bf16.h>

// CDNA5 / gfx1250 WMMA types (wave32)
typedef __attribute__((ext_vector_type(16))) __bf16 v16bf;
typedef __attribute__((ext_vector_type(8)))  float  v8f;

#define VOCAB 32000
#define EMB   512
#define HID   512
#define BSZ   32
#define TLEN  32
#define G4H   2048          // 4*HID
#define START_TOK 1
#define STOP_TOK  2

// ---------- helpers ----------
__device__ __forceinline__ unsigned int pack_bf16x2(float a, float b) {
  unsigned int ua = __float_as_uint(a);
  unsigned int ub = __float_as_uint(b);
  unsigned int ra = (ua + 0x7FFFu + ((ua >> 16) & 1u)) >> 16;   // RNE
  unsigned int rb = (ub + 0x7FFFu + ((ub >> 16) & 1u)) >> 16;
  return (ra & 0xFFFFu) | (rb << 16);
}

// ---------- weight packing: row-major fp32 [K,N] -> per-tile bf16 B-fragments ----------
// Tile (nt,kt) covers N=[nt*16,..), K=[kt*32,..). Tiles stored [nt][kt], 1024B each.
// Within a tile, lane l owns 32 bytes at l*32: dword j = (W[k,n], W[k+1,n]) with
// n = nt*16 + (l&15), k = kt*32 + (l>>4)*8 + 2*j for j<4, +16 for j>=4 (ISA 7.12.2, 16-bit B 32x16).
__global__ void pack_weight_bf16(const float* __restrict__ W,
                                 unsigned int* __restrict__ dst, int K, int N) {
  int d = blockIdx.x * blockDim.x + threadIdx.x;
  int total = (K * N) >> 1;
  if (d >= total) return;
  int KT = K >> 5;
  int t    = d >> 8;
  int nt   = t / KT;
  int kt   = t - nt * KT;
  int lane = (d >> 3) & 31;
  int j    = d & 7;
  int n = nt * 16 + (lane & 15);
  int k = kt * 32 + ((j < 4) ? (((lane >> 4) << 3) + (j << 1)) : (16 + ((lane >> 4) << 3) + ((j - 4) << 1)));
  dst[d] = pack_bf16x2(W[(size_t)k * N + n], W[(size_t)(k + 1) * N + n]);
}

// ---------- activation packing: fp32 [32,512] -> A-fragments (16x32 bf16 tiles) ----------
// Tiles stored [mt][kt] (mt in 0..1, kt in 0..15), 1024B each.
// Lane l: m = mt*16 + (l&15); dword j<4: k = kt*32 + (l>>4)*8 + 2j ; j>=4: +16.
__global__ void pack_act_bf16(const float* __restrict__ act,
                              unsigned int* __restrict__ dst) {
  int d = blockIdx.x * blockDim.x + threadIdx.x;       // 8192 dwords total
  if (d >= (BSZ * HID) / 2) return;
  int t    = d >> 8;
  int mt   = t >> 4;
  int kt   = t & 15;
  int lane = (d >> 3) & 31;
  int j    = d & 7;
  int m     = mt * 16 + (lane & 15);
  int khalf = lane >> 4;
  int k = kt * 32 + ((j < 4) ? (khalf * 8 + 2 * j) : (16 + khalf * 8 + 2 * (j - 4)));
  dst[d] = pack_bf16x2(act[m * HID + k], act[m * HID + k + 1]);
}

// Same, but rows gathered from the embedding table by token id.
__global__ void pack_emb_bf16(const float* __restrict__ emb,
                              const int* __restrict__ toks,
                              unsigned int* __restrict__ dst) {
  int d = blockIdx.x * blockDim.x + threadIdx.x;
  if (d >= (BSZ * EMB) / 2) return;
  int t    = d >> 8;
  int mt   = t >> 4;
  int kt   = t & 15;
  int lane = (d >> 3) & 31;
  int j    = d & 7;
  int m     = mt * 16 + (lane & 15);
  int khalf = lane >> 4;
  int k = kt * 32 + ((j < 4) ? (khalf * 8 + 2 * j) : (16 + khalf * 8 + 2 * (j - 4)));
  const float* row = emb + (size_t)toks[m] * EMB;
  dst[d] = pack_bf16x2(row[k], row[k + 1]);
}

// ---------- state init ----------
__global__ void init_state(const float* __restrict__ x, float* __restrict__ h,
                           float* __restrict__ c, int* __restrict__ toks,
                           float* __restrict__ loss) {
  int i = blockIdx.x * blockDim.x + threadIdx.x;
  if (i < BSZ * HID) { h[i] = x[i]; c[i] = 0.0f; }
  if (i < BSZ)       { toks[i] = START_TOK; loss[i] = 0.0f; }
}

// ---------- gates = inp@Wi + h@Wh  (bf16 WMMA, f32 accumulate) ----------
// One wave computes a full 32x16 strip (both m-tiles) so each weight fragment
// is loaded once and reused by two WMMAs. 128 n-tiles -> 16 blocks x 8 waves.
__global__ void gates_gemm(const unsigned short* __restrict__ Apx,
                           const unsigned short* __restrict__ Aph,
                           const unsigned short* __restrict__ Wip,
                           const unsigned short* __restrict__ Whp,
                           float* __restrict__ gates) {
  const int KT = EMB / 32;                         // 16
  int nt   = blockIdx.x * 8 + (threadIdx.x >> 5);  // 0..127
  int lane = threadIdx.x & 31;
  v8f acc0 = {};   // m-tile 0 (rows 0..15)
  v8f acc1 = {};   // m-tile 1 (rows 16..31)
  for (int kt = 0; kt < KT; ++kt) {
    if (kt + 2 < KT) {
      __builtin_prefetch(Wip + ((size_t)(nt * KT + kt + 2)) * 512 + lane * 16, 0, 1);
      __builtin_prefetch(Whp + ((size_t)(nt * KT + kt + 2)) * 512 + lane * 16, 0, 1);
    }
    v16bf wi  = *(const v16bf*)(Wip + (size_t)(nt * KT + kt) * 512 + lane * 16);
    v16bf ax0 = *(const v16bf*)(Apx + (size_t)(0 * KT + kt) * 512 + lane * 16);
    v16bf ax1 = *(const v16bf*)(Apx + (size_t)(1 * KT + kt) * 512 + lane * 16);
    acc0 = __builtin_amdgcn_wmma_f32_16x16x32_bf16(false, ax0, false, wi,
                                                   (short)0, acc0, false, false);
    acc1 = __builtin_amdgcn_wmma_f32_16x16x32_bf16(false, ax1, false, wi,
                                                   (short)0, acc1, false, false);
    v16bf wh  = *(const v16bf*)(Whp + (size_t)(nt * KT + kt) * 512 + lane * 16);
    v16bf ah0 = *(const v16bf*)(Aph + (size_t)(0 * KT + kt) * 512 + lane * 16);
    v16bf ah1 = *(const v16bf*)(Aph + (size_t)(1 * KT + kt) * 512 + lane * 16);
    acc0 = __builtin_amdgcn_wmma_f32_16x16x32_bf16(false, ah0, false, wh,
                                                   (short)0, acc0, false, false);
    acc1 = __builtin_amdgcn_wmma_f32_16x16x32_bf16(false, ah1, false, wh,
                                                   (short)0, acc1, false, false);
  }
  // C/D layout: VGPR i -> M = i + 8*(lane>=16), N = lane&15   (ISA 7.12.2)
  int n     = nt * 16 + (lane & 15);
  int mbase = (lane >> 4) << 3;
#pragma unroll
  for (int i = 0; i < 8; ++i) {
    gates[(size_t)(mbase + i)      * G4H + n] = acc0[i];
    gates[(size_t)(mbase + i + 16) * G4H + n] = acc1[i];
  }
}

// ---------- LSTM pointwise (gate order i,f,g,o like PyTorch) ----------
__global__ void lstm_pointwise(const float* __restrict__ gates,
                               const float* __restrict__ b,
                               float* __restrict__ c, float* __restrict__ h) {
  int idx = blockIdx.x * blockDim.x + threadIdx.x;
  if (idx >= BSZ * HID) return;
  int m = idx >> 9;
  int j = idx & 511;
  const float* g = gates + (size_t)m * G4H;
  float i_ = g[j]            + b[j];
  float f_ = g[HID + j]      + b[HID + j];
  float g_ = g[2 * HID + j]  + b[2 * HID + j];
  float o_ = g[3 * HID + j]  + b[3 * HID + j];
  float si = 1.0f / (1.0f + __expf(-i_));
  float sf = 1.0f / (1.0f + __expf(-f_));
  float so = 1.0f / (1.0f + __expf(-o_));
  float cn = sf * c[idx] + si * tanhf(g_);
  c[idx] = cn;
  h[idx] = so * tanhf(cn);
}

// ---------- logits = h@Wo  (bf16 WMMA) ----------
// One wave per n-tile computes the full 32x16 strip: the Wo fragment (the
// dominant L2 stream) is loaded once and reused by both m-tile WMMAs.
// 2000 n-tiles -> 250 blocks x 8 waves.
__global__ void out_gemm(const unsigned short* __restrict__ Aph,
                         const unsigned short* __restrict__ Wop,
                         float* __restrict__ logits) {
  const int KT = HID / 32;                          // 16
  int nt   = blockIdx.x * 8 + (threadIdx.x >> 5);   // 0..1999
  int lane = threadIdx.x & 31;
  v8f acc0 = {};
  v8f acc1 = {};
  for (int kt = 0; kt < KT; ++kt) {
    if (kt + 2 < KT)
      __builtin_prefetch(Wop + ((size_t)(nt * KT + kt + 2)) * 512 + lane * 16, 0, 1);
    v16bf w  = *(const v16bf*)(Wop + (size_t)(nt * KT + kt) * 512 + lane * 16);
    v16bf a0 = *(const v16bf*)(Aph + (size_t)(0 * KT + kt) * 512 + lane * 16);
    v16bf a1 = *(const v16bf*)(Aph + (size_t)(1 * KT + kt) * 512 + lane * 16);
    acc0 = __builtin_amdgcn_wmma_f32_16x16x32_bf16(false, a0, false, w,
                                                   (short)0, acc0, false, false);
    acc1 = __builtin_amdgcn_wmma_f32_16x16x32_bf16(false, a1, false, w,
                                                   (short)0, acc1, false, false);
  }
  int n     = nt * 16 + (lane & 15);
  int mbase = (lane >> 4) << 3;
#pragma unroll
  for (int i = 0; i < 8; ++i) {
    logits[(size_t)(mbase + i)      * VOCAB + n] = acc0[i];
    logits[(size_t)(mbase + i + 16) * VOCAB + n] = acc1[i];
  }
}

// ---------- fused log-softmax + CE + argmax + scheduled sampling ----------
// One block (1024 threads = 32 waves) per batch row. loss[m] updated only by
// block m -> no atomics, fully deterministic across replays.
__global__ void softmax_ce(const float* __restrict__ logits,
                           const float* __restrict__ bo,
                           const int* __restrict__ labels,
                           const unsigned char* __restrict__ coin,
                           int t, int stop_mode,
                           float* __restrict__ loss, int* __restrict__ toks) {
  __shared__ float sval[1024];
  __shared__ int   sidx[1024];
  int m   = blockIdx.x;
  int tid = threadIdx.x;
  const float* row = logits + (size_t)m * VOCAB;

  // pass 1: max + argmax (first index on ties)
  float best = -INFINITY;
  int   bidx = VOCAB;
  for (int i = tid; i < VOCAB; i += 1024) {
    float v = row[i] + bo[i];
    if (v > best) { best = v; bidx = i; }
  }
  sval[tid] = best; sidx[tid] = bidx;
  __syncthreads();
  for (int s = 512; s > 0; s >>= 1) {
    if (tid < s) {
      float v = sval[tid + s]; int ix = sidx[tid + s];
      if (v > sval[tid] || (v == sval[tid] && ix < sidx[tid])) {
        sval[tid] = v; sidx[tid] = ix;
      }
    }
    __syncthreads();
  }
  float vmax = sval[0];
  int   amax = sidx[0];
  __syncthreads();

  // pass 2: sum of exp(x - max)
  float sum = 0.0f;
  for (int i = tid; i < VOCAB; i += 1024) sum += __expf(row[i] + bo[i] - vmax);
  sval[tid] = sum;
  __syncthreads();
  for (int s = 512; s > 0; s >>= 1) {
    if (tid < s) sval[tid] += sval[tid + s];
    __syncthreads();
  }

  if (tid == 0) {
    int   label = stop_mode ? STOP_TOK : labels[m * TLEN + t];
    float lx    = row[label] + bo[label];
    float ce    = -(lx - vmax - __logf(sval[0]));
    loss[m] += ce;
    if (!stop_mode) {
      toks[m] = (coin[m * TLEN + t] != 0) ? amax : label;
    }
  }
}

__global__ void finalize(const float* __restrict__ loss, float* __restrict__ out) {
  if (blockIdx.x == 0 && threadIdx.x == 0) {
    float s = 0.0f;
    for (int i = 0; i < BSZ; ++i) s += loss[i];
    out[0] = s / (float)BSZ;
  }
}

// ---------- host ----------
extern "C" void kernel_launch(void* const* d_in, const int* in_sizes, int n_in,
                              void* d_out, int out_size, void* d_ws, size_t ws_size,
                              hipStream_t stream) {
  (void)in_sizes; (void)n_in; (void)out_size; (void)ws_size;
  const float*         x      = (const float*)d_in[0];
  const int*           labels = (const int*)d_in[1];
  const unsigned char* coin   = (const unsigned char*)d_in[2];  // jnp.bool_: 1 byte
  const float*         emb    = (const float*)d_in[3];
  const float*         Wi     = (const float*)d_in[4];
  const float*         Wh     = (const float*)d_in[5];
  const float*         b      = (const float*)d_in[6];
  const float*         Wo     = (const float*)d_in[7];
  const float*         bo     = (const float*)d_in[8];

  char* p = (char*)d_ws;
  auto alloc = [&](size_t bytes) {
    char* r = p; p += (bytes + 255) & ~((size_t)255); return r;
  };
  unsigned short* Wip    = (unsigned short*)alloc((size_t)EMB * G4H * 2);   //  2 MB
  unsigned short* Whp    = (unsigned short*)alloc((size_t)HID * G4H * 2);   //  2 MB
  unsigned short* Wop    = (unsigned short*)alloc((size_t)HID * VOCAB * 2); // 32 MB
  unsigned short* Apx    = (unsigned short*)alloc((size_t)BSZ * EMB * 2);
  unsigned short* Aph    = (unsigned short*)alloc((size_t)BSZ * HID * 2);
  float*          gates  = (float*)alloc((size_t)BSZ * G4H * 4);
  float*          hbuf   = (float*)alloc((size_t)BSZ * HID * 4);
  float*          cbuf   = (float*)alloc((size_t)BSZ * HID * 4);
  float*          logits = (float*)alloc((size_t)BSZ * VOCAB * 4);          //  4 MB
  float*          loss   = (float*)alloc(256);
  int*            toks   = (int*)alloc(256);

  // One-time bf16 weight packing into WMMA fragment layout (L2-resident after).
  {
    int tot = EMB * G4H / 2;
    pack_weight_bf16<<<(tot + 255) / 256, 256, 0, stream>>>(Wi, (unsigned int*)Wip, EMB, G4H);
    pack_weight_bf16<<<(tot + 255) / 256, 256, 0, stream>>>(Wh, (unsigned int*)Whp, HID, G4H);
    int tot2 = HID * VOCAB / 2;
    pack_weight_bf16<<<(tot2 + 255) / 256, 256, 0, stream>>>(Wo, (unsigned int*)Wop, HID, VOCAB);
  }

  init_state<<<(BSZ * HID + 255) / 256, 256, 0, stream>>>(x, hbuf, cbuf, toks, loss);
  pack_act_bf16<<<32, 256, 0, stream>>>(hbuf, (unsigned int*)Aph);

  for (int t = 0; t <= TLEN; ++t) {
    int stop = (t == TLEN) ? 1 : 0;
    pack_emb_bf16<<<32, 256, 0, stream>>>(emb, toks, (unsigned int*)Apx);
    gates_gemm<<<16, 256, 0, stream>>>(Apx, Aph, Wip, Whp, gates);
    lstm_pointwise<<<64, 256, 0, stream>>>(gates, b, cbuf, hbuf);
    pack_act_bf16<<<32, 256, 0, stream>>>(hbuf, (unsigned int*)Aph);
    out_gemm<<<250, 256, 0, stream>>>(Aph, Wop, logits);
    softmax_ce<<<BSZ, 1024, 0, stream>>>(logits, bo, labels, coin, t, stop, loss, toks);
  }

  finalize<<<1, 32, 0, stream>>>(loss, (float*)d_out);
}
